// W8A8Linear_24489903521827
// MI455X (gfx1250) — 8-bit matrix operand order repaired and checked by reading, compiled, not run on hardware
//
#include <hip/hip_runtime.h>
#include <stdint.h>

typedef int v8i __attribute__((ext_vector_type(8)));

#define M_DIM 8192
#define N_DIM 4096
#define K_DIM 4096

__global__ __launch_bounds__(256) void absmax_kernel(const float4* __restrict__ v,
                                                     long n4,
                                                     unsigned* __restrict__ out) {
    float m = 0.0f;
    for (long i = (long)blockIdx.x * blockDim.x + threadIdx.x; i < n4;
         i += (long)gridDim.x * blockDim.x) {
        float4 t = v[i];
        m = fmaxf(m, fmaxf(fmaxf(fabsf(t.x), fabsf(t.y)),
                           fmaxf(fabsf(t.z), fabsf(t.w))));
    }
    __shared__ float red[256];
    red[threadIdx.x] = m;
    __syncthreads();
    for (int s = 128; s > 0; s >>= 1) {
        if (threadIdx.x < (unsigned)s)
            red[threadIdx.x] = fmaxf(red[threadIdx.x], red[threadIdx.x + s]);
        __syncthreads();
    }
    if (threadIdx.x == 0) atomicMax(out, __float_as_uint(red[0]));
}

__global__ __launch_bounds__(256) void quant_kernel(const float4* __restrict__ v,
                                                    long n4,
                                                    const unsigned* __restrict__ am_bits,
                                                    int* __restrict__ q) {
    const float am  = __uint_as_float(*am_bits);
    const float inv = 127.0f / fmaxf(am, 1e-5f);
    for (long i = (long)blockIdx.x * blockDim.x + threadIdx.x; i < n4;
         i += (long)gridDim.x * blockDim.x) {
        float4 t = v[i];
        int b0 = __float2int_rn(t.x * inv);
        int b1 = __float2int_rn(t.y * inv);
        int b2 = __float2int_rn(t.z * inv);
        int b3 = __float2int_rn(t.w * inv);
        q[i] = (b0 & 0xff) | ((b1 & 0xff) << 8) | ((b2 & 0xff) << 16)
             | ((b3 & 0xff) << 24);
    }
}

__global__ __launch_bounds__(256) void gemm_iu8_kernel(const int8_t* __restrict__ Aq,
                                                       const int8_t* __restrict__ Bq,
                                                       const unsigned* __restrict__ am,
                                                       const float* __restrict__ bias,
                                                       float* __restrict__ out) {
    __shared__ __align__(16) char Bs[2][64 * 64];

    const int tid  = threadIdx.x;
    const int lane = tid & 31;
    const int wave = tid >> 5;
    const int half = lane >> 4;
    const int l16  = lane & 15;

    const int n_base = blockIdx.x * 64;
    const int m_base = blockIdx.y * 256 + wave * 32;

    const int sn = tid >> 2;
    const int sc = tid & 3;
    const int8_t* bsrc  = Bq + (size_t)(n_base + sn) * K_DIM + sc * 16;
    const unsigned lds0 = (unsigned)(uintptr_t)(&Bs[0][sn * 64 + sc * 16]);
    const unsigned lds1 = (unsigned)(uintptr_t)(&Bs[1][sn * 64 + sc * 16]);

    v8i acc[2][4] = {{{}, {}, {}, {}}, {{}, {}, {}, {}}};

    asm volatile("global_load_async_to_lds_b128 %0, %1, off"
                 :: "v"(lds0), "v"(bsrc) : "memory");
    asm volatile("s_wait_asynccnt 0" ::: "memory");
    __syncthreads();

    for (int k0 = 0; k0 < K_DIM; k0 += 64) {
        const int cur = (k0 >> 6) & 1;

        if (k0 + 64 < K_DIM) {
            const unsigned dst = cur ? lds0 : lds1;
            asm volatile("global_load_async_to_lds_b128 %0, %1, off"
                         :: "v"(dst), "v"(bsrc + k0 + 64) : "memory");
        }

        v8i a0, a1;
        {
            const int8_t* arow0 = Aq + (size_t)(m_base + l16) * K_DIM + k0 + half * 16;
            const int8_t* arow1 = arow0 + (size_t)16 * K_DIM;
#pragma unroll
            for (int g = 0; g < 4; ++g) {
                const int* p0 = (const int*)(arow0 + (g & 1) * 8 + (g >> 1) * 32);
                const int* p1 = (const int*)(arow1 + (g & 1) * 8 + (g >> 1) * 32);
                a0[2 * g]     = p0[0];
                a0[2 * g + 1] = p0[1];
                a1[2 * g]     = p1[0];
                a1[2 * g + 1] = p1[1];
            }
        }

        v8i b[4];
#pragma unroll
        for (int nt = 0; nt < 4; ++nt) {
            const char* brow = &Bs[cur][(nt * 16 + l16) * 64 + half * 16];
            const int4 x0 = *(const int4*)(brow);
            const int4 x1 = *(const int4*)(brow + 32);
            b[nt] = (v8i){x0.x, x0.y, x0.z, x0.w, x1.x, x1.y, x1.z, x1.w};
        }

#pragma unroll
        for (int nt = 0; nt < 4; ++nt) {
            acc[0][nt] = __builtin_amdgcn_wmma_i32_16x16x64_iu8(
                true, a0, true, b[nt], acc[0][nt], false, false);
            acc[1][nt] = __builtin_amdgcn_wmma_i32_16x16x64_iu8(
                true, a1, true, b[nt], acc[1][nt], false, false);
        }

        asm volatile("s_wait_asynccnt 0" ::: "memory");
        __syncthreads();
    }

    const float ax = __uint_as_float(am[0]);
    const float aw = __uint_as_float(am[1]);
    const float s  = (fmaxf(ax, 1e-5f) / 127.0f) * (fmaxf(aw, 1e-5f) / 127.0f);

#pragma unroll
    for (int mt = 0; mt < 2; ++mt) {
#pragma unroll
        for (int nt = 0; nt < 4; ++nt) {
            const int   n  = n_base + nt * 16 + l16;
            const float bv = bias[n];
#pragma unroll
            for (int r = 0; r < 8; ++r) {
                const int m = m_base + mt * 16 + r + 8 * half;
                out[(size_t)m * N_DIM + n] = (float)acc[mt][nt][r] * s + bv;
            }
        }
    }
}

extern "C" void kernel_launch(void* const* d_in, const int* in_sizes, int n_in,
                              void* d_out, int out_size, void* d_ws, size_t ws_size,
                              hipStream_t stream) {
    const float* x    = (const float*)d_in[0];
    const float* w    = (const float*)d_in[1];
    const float* bias = (const float*)d_in[2];
    float*       out  = (float*)d_out;

    unsigned* am = (unsigned*)d_ws;
    int8_t*   Aq = (int8_t*)d_ws + 256;
    int8_t*   Bq = Aq + (size_t)M_DIM * K_DIM;

    const long nx = (long)M_DIM * K_DIM;
    const long nw = (long)N_DIM * K_DIM;

    hipMemsetAsync(am, 0, 2 * sizeof(unsigned), stream);

    absmax_kernel<<<2048, 256, 0, stream>>>((const float4*)x, nx / 4, am + 0);
    absmax_kernel<<<2048, 256, 0, stream>>>((const float4*)w, nw / 4, am + 1);

    quant_kernel<<<4096, 256, 0, stream>>>((const float4*)x, nx / 4, am + 0, (int*)Aq);
    quant_kernel<<<4096, 256, 0, stream>>>((const float4*)w, nw / 4, am + 1, (int*)Bq);

    dim3 grid(N_DIM / 64, M_DIM / 256);
    gemm_iu8_kernel<<<grid, 256, 0, stream>>>(Aq, Bq, am, bias, out);
}
